// GuardedTriXGR6502_18391049961870
// MI455X (gfx1250) — compile-verified, matrix operand order checked by reading, not run on hardware
//
#include <hip/hip_runtime.h>
#include <hip/hip_bf16.h>
#include <math.h>

// ---------------------------------------------------------------------------
// Problem constants (must match reference)
// ---------------------------------------------------------------------------
#define BSZ   4096
#define DMOD  256
#define NT    16
#define DFF   1024
#define TERNARY_W   0.01f
#define SPARSITY_W  0.005f
#define DIVERSITY_W 0.01f

// fragment geometry
#define KK1   (DMOD / 32)        // 8  K-steps for phase 1
#define NT1   (DFF / 16)         // 64 feature-tiles for phase 1
#define KK2   (DFF / 32)         // 32 K-steps for phase 2
#define NT2   (DMOD / 16)        // 16 feature-tiles for phase 2
#define FRAG_HALVES 512          // 32 lanes * 16 halves
#define NFRAG1 (NT * KK1 * NT1)  // 8192
#define NFRAG2 (NT * KK2 * NT2)  // 8192

typedef __attribute__((ext_vector_type(16))) _Float16 v16h;
typedef __attribute__((ext_vector_type(8)))  _Float16 h8;
typedef __attribute__((ext_vector_type(8)))  float    v8f;

// ---------------------------------------------------------------------------
// Workspace layout (bytes, 256B aligned slots)
// ---------------------------------------------------------------------------
constexpr size_t OFF_XH     = 0;                                     // f16 [B, D]
constexpr size_t OFF_GATES  = OFF_XH     + (size_t)2 * BSZ * DMOD;   // f32 [B]
constexpr size_t OFF_SIG    = OFF_GATES  + (size_t)4 * BSZ;          // f32 [T, D]
constexpr size_t OFF_COUNTS = OFF_SIG    + (size_t)4 * NT * DMOD;    // i32 [T]
constexpr size_t OFF_PMEAN  = OFF_COUNTS + 256;                      // f32 [T]
constexpr size_t OFF_AUXP   = OFF_PMEAN  + 256;                      // f32 [1]
constexpr size_t OFF_BUCKET = OFF_AUXP   + 256;                      // i32 [T, B]
constexpr size_t OFF_EO     = OFF_BUCKET + (size_t)4 * NT * BSZ;     // f32 [B, D]
constexpr size_t OFF_PW1    = OFF_EO     + (size_t)4 * BSZ * DMOD;   // f16 packed w1
constexpr size_t OFF_PW2    = OFF_PW1    + (size_t)2 * NFRAG1 * FRAG_HALVES;
// total = OFF_PW2 + 2*NFRAG2*512 bytes  ~= 23.4 MB

// ===========================================================================
// Kernel P: pre-pack w1/w2 (f32, row-major) into WMMA fragment layout (f16).
// Fragment (kk, nt) per lane: 16 contiguous halves (32B), W[k][n] with
// n = nt*16 + (lane&15) and k runs per the standard f16 A/B lane mapping:
//   lane<16 : elems 0..7 -> K=kb+0..7,  8..15 -> K=kb+16..23
//   lane>=16: elems 0..7 -> K=kb+8..15, 8..15 -> K=kb+24..31
// Used as the *A* operand (M = feature axis) in the transposed GEMM.
// ===========================================================================
__global__ __launch_bounds__(256) void pack_w_kernel(
    const float* __restrict__ w1, const float* __restrict__ w2,
    _Float16* __restrict__ pw1, _Float16* __restrict__ pw2)
{
    const int wave = blockIdx.x * 8 + (threadIdx.x >> 5);
    const int lane = threadIdx.x & 31;
    const int hif  = (lane >> 4) & 1;
    const int ncol = lane & 15;

    if (wave < NFRAG1) {
        const int f  = wave;
        const int e  = f >> 9;            // / (KK1*NT1)
        const int kk = (f >> 6) & 7;
        const int nt = f & 63;
        const float* src = w1 + (size_t)e * DMOD * DFF + (size_t)(nt * 16 + ncol);
        const int kb = kk * 32 + hif * 8;
        _Float16* dst = pw1 + (size_t)f * FRAG_HALVES + (size_t)lane * 16;
        h8 lo, hi;
#pragma unroll
        for (int i = 0; i < 8; ++i) {
            lo[i] = (_Float16)src[(size_t)(kb + i) * DFF];
            hi[i] = (_Float16)src[(size_t)(kb + 16 + i) * DFF];
        }
        *(h8*)(dst)     = lo;
        *(h8*)(dst + 8) = hi;
    } else {
        const int f  = wave - NFRAG1;
        const int e  = f >> 9;            // / (KK2*NT2)
        const int kk = (f >> 4) & 31;
        const int nt = f & 15;
        const float* src = w2 + (size_t)e * DFF * DMOD + (size_t)(nt * 16 + ncol);
        const int kb = kk * 32 + hif * 8;
        _Float16* dst = pw2 + (size_t)f * FRAG_HALVES + (size_t)lane * 16;
        h8 lo, hi;
#pragma unroll
        for (int i = 0; i < 8; ++i) {
            lo[i] = (_Float16)src[(size_t)(kb + i) * DMOD];
            hi[i] = (_Float16)src[(size_t)(kb + 16 + i) * DMOD];
        }
        *(h8*)(dst)     = lo;
        *(h8*)(dst + 8) = hi;
    }
}

// ===========================================================================
// Kernel 0: signatures = tanh(sig); ternary + diversity aux; zero counters.
// ===========================================================================
__global__ __launch_bounds__(256) void setup_kernel(
    const float* __restrict__ sig, float* __restrict__ sigbuf,
    int* __restrict__ counts, float* __restrict__ pmean,
    float* __restrict__ auxp)
{
    const int tid = threadIdx.x;
    __shared__ float sSig[NT * DMOD];   // 16 KB
    __shared__ float red[256];
    __shared__ float norms[NT];
    __shared__ float sTern;

    float tern = 0.f;
    for (int i = tid; i < NT * DMOD; i += 256) {
        float s = tanhf(sig[i]);
        sSig[i]   = s;
        sigbuf[i] = s;
        float ab = fabsf(s);
        tern += ab * (1.f - ab);
    }
    red[tid] = tern;
    __syncthreads();
    for (int off = 128; off > 0; off >>= 1) {
        if (tid < off) red[tid] += red[tid + off];
        __syncthreads();
    }
    if (tid == 0) sTern = TERNARY_W * red[0] * (1.f / (NT * DMOD));
    if (tid < NT) {
        float ss = 0.f;
        const float* row = &sSig[tid * DMOD];
        for (int d = 0; d < DMOD; ++d) ss += row[d] * row[d];
        norms[tid] = sqrtf(ss) + 1e-6f;
    }
    __syncthreads();
    {
        const int i = tid >> 4, j = tid & 15;
        const float* ri = &sSig[i * DMOD];
        const float* rj = &sSig[j * DMOD];
        float g = 0.f;
        for (int d = 0; d < DMOD; ++d) g += ri[d] * rj[d];
        g /= (norms[i] * norms[j]);
        float dd = g - (i == j ? 1.f : 0.f);
        red[tid] = dd * dd;
    }
    __syncthreads();
    for (int off = 128; off > 0; off >>= 1) {
        if (tid < off) red[tid] += red[tid + off];
        __syncthreads();
    }
    if (tid == 0) auxp[0] = sTern + DIVERSITY_W * red[0] * (1.f / (NT * NT));
    if (tid < NT) { counts[tid] = 0; pmean[tid] = 0.f; }
}

// ===========================================================================
// Kernel A: front-end per token. 4096 blocks x 256 threads.
// ===========================================================================
__global__ __launch_bounds__(256) void frontend_kernel(
    const int* __restrict__ op_idx, const int* __restrict__ ai,
    const int* __restrict__ bi, const int* __restrict__ ci,
    const float* __restrict__ op_table, const float* __restrict__ proj_w,
    const float* __restrict__ proj_b, const float* __restrict__ mix_w,
    const float* __restrict__ mix_b, const float* __restrict__ sigbuf,
    _Float16* __restrict__ xh, float* __restrict__ gates,
    int* __restrict__ counts, float* __restrict__ pmean,
    int* __restrict__ bucket, float* __restrict__ tile_out)
{
    const int b   = blockIdx.x;
    const int tid = threadIdx.x;
    __shared__ float s49[64];
    __shared__ float sT[DMOD];
    __shared__ float sX[DMOD];
    __shared__ float sS[NT];
    __shared__ float sE[NT];
    __shared__ float sMax, sSum;
    __shared__ int   sArg;

    if (tid < 32)       s49[tid] = op_table[op_idx[b] * 32 + tid];
    else if (tid < 40)  s49[tid] = (float)((ai[b] >> (tid - 32)) & 1);
    else if (tid < 48)  s49[tid] = (float)((bi[b] >> (tid - 40)) & 1);
    else if (tid == 48) s49[48]  = (float)ci[b];
    __syncthreads();

    float x = proj_b[tid];
    for (int k = 0; k < 49; ++k) x += s49[k] * proj_w[k * DMOD + tid];
    sT[tid] = tanhf(x);
    __syncthreads();

    float mv = mix_b[tid];
    for (int k = 0; k < DMOD; ++k) mv += sT[k] * mix_w[k * DMOD + tid];
    x += mv;
    sX[tid] = x;
    xh[(size_t)b * DMOD + tid] = (_Float16)x;
    __syncthreads();

    if (tid < NT) {
        const float* sg = sigbuf + tid * DMOD;
        float sc = 0.f;
        for (int d = 0; d < DMOD; ++d) sc += sX[d] * sg[d];
        sS[tid] = sc;
    }
    __syncthreads();
    if (tid == 0) {
        float mx = sS[0]; int am = 0;
        for (int t = 1; t < NT; ++t) if (sS[t] > mx) { mx = sS[t]; am = t; }
        sMax = mx; sArg = am;
    }
    __syncthreads();
    if (tid < NT) sE[tid] = expf(sS[tid] - sMax);
    __syncthreads();
    if (tid == 0) {
        float s = 0.f;
        for (int t = 0; t < NT; ++t) s += sE[t];
        sSum = s;
    }
    __syncthreads();
    if (tid < NT) atomicAdd(&pmean[tid], sE[tid] / sSum);
    if (tid == 0) {
        const int am = sArg;
        gates[b]    = sE[am] / sSum;
        tile_out[b] = (float)am;
        int pos = atomicAdd(&counts[am], 1);
        bucket[am * BSZ + pos] = b;
    }
}

// ===========================================================================
// Kernel C: routed expert FFN, *transposed* WMMA:  h^T = W1^T x X^T.
//   A operand = packed weight fragment  (M = feature axis)
//   B operand = activation fragment     (N = token axis, token = lane&15)
// D layout then gives each lane 8 CONTIGUOUS features of one token ->
// vectorized b128 LDS stores (phase 1) and 32B global stores (phase 2).
// grid = (B/32, T); 256 threads = 8 waves; 32 tokens/block.
// ===========================================================================
__global__ __launch_bounds__(256) void expert_ffn_kernel(
    const _Float16* __restrict__ xh, const _Float16* __restrict__ pw1,
    const float* __restrict__ b1, const _Float16* __restrict__ pw2,
    const float* __restrict__ b2, const int* __restrict__ bucket,
    const int* __restrict__ counts, float* __restrict__ eo)
{
    const int e   = blockIdx.y;
    const int cnt = counts[e];
    const int m0  = blockIdx.x * 32;
    if (m0 >= cnt) return;

    __shared__ __align__(16) _Float16 sH[32][1040];  // 66.5 KB, padded stride
    __shared__ int tok[32];

    const int tid = threadIdx.x;
    if (tid < 32) {
        int idx = m0 + tid;
        tok[tid] = bucket[e * BSZ + (idx < cnt ? idx : m0)];
    }
    __syncthreads();

    const int lane = tid & 31;
    const int w    = tid >> 5;
    const int mt   = w & 1;            // token tile (N side of WMMA)
    const int grp  = w >> 1;           // feature group
    const int tcol = lane & 15;        // token within tile (B-frag column)
    const int hif  = (lane >> 4) & 1;  // lane half -> K sub-block / feature half
    const int m    = mt * 16 + tcol;   // token row in block

    // -------- preload B fragments (X^T) for all 8 K-steps (K=256) ----------
    const _Float16* xr = xh + (size_t)tok[m] * DMOD;
    v16h bx[8];
#pragma unroll
    for (int kk = 0; kk < 8; ++kk) {
        const _Float16* p = xr + kk * 32 + hif * 8;
        h8 lo = *(const h8*)(p);
        h8 hb = *(const h8*)(p + 16);
#pragma unroll
        for (int i = 0; i < 8; ++i) { bx[kk][i] = lo[i]; bx[kk][8 + i] = hb[i]; }
    }

    // -------- phase 1: h^T = W1^T @ X^T (+b1, relu), 16 feature tiles ------
    const _Float16* p1base = pw1 + (size_t)e * KK1 * NT1 * FRAG_HALVES
                                 + (size_t)lane * 16;
    for (int nt = 0; nt < 16; ++nt) {
        const int ntg = grp * 16 + nt;
        v8f acc = {};
#pragma unroll
        for (int kk = 0; kk < 8; ++kk) {
            v16h aw = *(const v16h*)(p1base + (size_t)(kk * NT1 + ntg) * FRAG_HALVES);
            acc = __builtin_amdgcn_wmma_f32_16x16x32_f16(
                false, aw, false, bx[kk], (short)0, acc, false, false);
        }
        // lane holds features n0..n0+7 of token m -> one b128 store
        const int n0 = ntg * 16 + hif * 8;
        const v8f bias = *(const v8f*)(b1 + e * DFF + n0);
        h8 hv;
#pragma unroll
        for (int r = 0; r < 8; ++r) {
            float v = acc[r] + bias[r];
            hv[r] = (_Float16)(v > 0.f ? v : 0.f);
        }
        *(h8*)(&sH[m][n0]) = hv;
    }
    __syncthreads();

    // -------- phase 2: y^T = W2^T @ H^T (+b2), 4 feature tiles, K=1024 -----
    const _Float16* p2base = pw2 + (size_t)e * KK2 * NT2 * FRAG_HALVES
                                 + (size_t)lane * 16;
    const _Float16* hr = &sH[m][0];
    v8f acc2[4] = {};
    for (int kk = 0; kk < 32; ++kk) {
        const _Float16* p = hr + kk * 32 + hif * 8;
        h8 lo = *(const h8*)(p);
        h8 hb = *(const h8*)(p + 16);
        v16h bh;
#pragma unroll
        for (int i = 0; i < 8; ++i) { bh[i] = lo[i]; bh[8 + i] = hb[i]; }
#pragma unroll
        for (int j = 0; j < 4; ++j) {
            const int ntg = grp * 4 + j;
            v16h aw = *(const v16h*)(p2base + (size_t)(kk * NT2 + ntg) * FRAG_HALVES);
            acc2[j] = __builtin_amdgcn_wmma_f32_16x16x32_f16(
                false, aw, false, bh, (short)0, acc2[j], false, false);
        }
    }
    // per lane: token m, contiguous features -> vector stores
    const int R = (cnt - m0) < 32 ? (cnt - m0) : 32;
    if (m < R) {
        float* orow = eo + (size_t)tok[m] * DMOD;
#pragma unroll
        for (int j = 0; j < 4; ++j) {
            const int n0 = (grp * 4 + j) * 16 + hif * 8;
            const v8f bias = *(const v8f*)(b2 + e * DMOD + n0);
            v8f out;
#pragma unroll
            for (int r = 0; r < 8; ++r) out[r] = acc2[j][r] + bias[r];
            *(v8f*)(orow + n0) = out;
        }
    }
}

// ===========================================================================
// Kernel B: finalize aux scalar.
// ===========================================================================
__global__ void finalize_aux_kernel(const int* __restrict__ counts,
                                    const float* __restrict__ pmean,
                                    const float* __restrict__ auxp,
                                    float* __restrict__ out_aux)
{
    if (threadIdx.x == 0) {
        float s = 0.f;
        for (int t = 0; t < NT; ++t)
            s += ((float)counts[t] * (1.f / BSZ)) * (pmean[t] * (1.f / BSZ));
        out_aux[0] = auxp[0] + SPARSITY_W * (float)NT * s;
    }
}

// ===========================================================================
// Kernel D: result head. 4096 blocks x 64 threads.
// ===========================================================================
__global__ __launch_bounds__(64) void head_kernel(
    const float* __restrict__ eo, const float* __restrict__ gates,
    const float* __restrict__ hw1, const float* __restrict__ hb1,
    const float* __restrict__ hw2, const float* __restrict__ hb2,
    float* __restrict__ result)
{
    const int b = blockIdx.x;
    const int tid = threadIdx.x;
    __shared__ float sv[DMOD];
    __shared__ float sz[64];
    const float g = gates[b];
    for (int i = tid; i < DMOD; i += 64) sv[i] = g * eo[(size_t)b * DMOD + i];
    __syncthreads();
    float z = hb1[tid];
    for (int d = 0; d < DMOD; ++d) z += sv[d] * hw1[d * 64 + tid];
    sz[tid] = z > 0.f ? z : 0.f;
    __syncthreads();
    if (tid < 8) {
        float r = hb2[tid];
        for (int k = 0; k < 64; ++k) r += sz[k] * hw2[k * 8 + tid];
        result[b * 8 + tid] = 1.f / (1.f + expf(-r));
    }
}

// ===========================================================================
// Launch
// ===========================================================================
extern "C" void kernel_launch(void* const* d_in, const int* in_sizes, int n_in,
                              void* d_out, int out_size, void* d_ws, size_t ws_size,
                              hipStream_t stream) {
    const int*   op_idx   = (const int*)  d_in[0];
    const int*   a_in     = (const int*)  d_in[1];
    const int*   b_in     = (const int*)  d_in[2];
    const int*   c_in     = (const int*)  d_in[3];
    const float* op_table = (const float*)d_in[4];
    const float* proj_w   = (const float*)d_in[5];
    const float* proj_b   = (const float*)d_in[6];
    const float* mix_w    = (const float*)d_in[7];
    const float* mix_b    = (const float*)d_in[8];
    const float* sig      = (const float*)d_in[9];
    const float* w1       = (const float*)d_in[10];
    const float* b1       = (const float*)d_in[11];
    const float* w2       = (const float*)d_in[12];
    const float* b2       = (const float*)d_in[13];
    const float* hw1      = (const float*)d_in[14];
    const float* hb1      = (const float*)d_in[15];
    const float* hw2      = (const float*)d_in[16];
    const float* hb2      = (const float*)d_in[17];

    char* ws = (char*)d_ws;
    _Float16* xh     = (_Float16*)(ws + OFF_XH);
    float*    gates  = (float*)   (ws + OFF_GATES);
    float*    sigbuf = (float*)   (ws + OFF_SIG);
    int*      counts = (int*)     (ws + OFF_COUNTS);
    float*    pmean  = (float*)   (ws + OFF_PMEAN);
    float*    auxp   = (float*)   (ws + OFF_AUXP);
    int*      bucket = (int*)     (ws + OFF_BUCKET);
    float*    eo     = (float*)   (ws + OFF_EO);
    _Float16* pw1    = (_Float16*)(ws + OFF_PW1);
    _Float16* pw2    = (_Float16*)(ws + OFF_PW2);

    float* out_result = (float*)d_out;                  // [B, 8]
    float* out_tile   = out_result + (size_t)BSZ * 8;   // [B]
    float* out_aux    = out_tile + BSZ;                 // [1]

    // P) pack weights into WMMA fragment layout (f16), one wave/fragment
    pack_w_kernel<<<(NFRAG1 + NFRAG2) / 8, 256, 0, stream>>>(w1, w2, pw1, pw2);

    // 0) signatures + static aux terms + zero counters
    setup_kernel<<<1, 256, 0, stream>>>(sig, sigbuf, counts, pmean, auxp);

    // 1) per-token front-end + routing scatter
    frontend_kernel<<<BSZ, 256, 0, stream>>>(
        op_idx, a_in, b_in, c_in, op_table, proj_w, proj_b, mix_w, mix_b,
        sigbuf, xh, gates, counts, pmean, bucket, out_tile);

    // 2) routed expert FFN with transposed WMMA on packed fragments
    dim3 grid(BSZ / 32, NT);
    expert_ffn_kernel<<<grid, 256, 0, stream>>>(
        xh, pw1, b1, pw2, b2, bucket, counts, eo);

    // 3) aux scalar
    finalize_aux_kernel<<<1, 32, 0, stream>>>(counts, pmean, auxp, out_aux);

    // 4) result head
    head_kernel<<<BSZ, 64, 0, stream>>>(
        eo, gates, hw1, hb1, hw2, hb2, out_result);
}
